// BiRNNScratch_34196529611283
// MI455X (gfx1250) — matrix operand order repaired and checked by reading, not compiled, not run on hardware
//
#include <hip/hip_runtime.h>
#include <hip/hip_bf16.h>
#include <cstdint>
#include <cstddef>

// ---------------------------------------------------------------------------
// Bidirectional RNN for MI455X (gfx1250, wave32, WMMA bf16 16x16x32).
//   T=512 B=64 D=512 H=1024
//   out = [ outputs (T,B,2H) | f_H (B,H) | b_H (B,H) ]  (all f32)
//
// Data path: weights pre-transposed to bf16 (k-contiguous) once; all
// contiguous tile staging uses gfx1250 async-to-LDS (ASYNCcnt) copies.
// ---------------------------------------------------------------------------

#define T_SEQ 512
#define BB    64
#define DD    512
#define HH    1024

// LDS row strides in ushort units (multiple of 8 -> 16B-aligned b128 rows)
#define AS 40   // A tile: 64 rows x 32 k
#define BS 40   // B tile: 128 n-rows x 32 k (k-contiguous)

typedef __attribute__((ext_vector_type(16))) __bf16      v16bf;
typedef __attribute__((ext_vector_type(8)))  float        v8f;
typedef __attribute__((ext_vector_type(8)))  unsigned int v8u;

static __device__ __forceinline__ unsigned short f2bf(float f) {
  unsigned int u = __float_as_uint(f);
  unsigned int r = u + 0x7FFFu + ((u >> 16) & 1u);   // round-to-nearest-even
  return (unsigned short)(r >> 16);
}

// gfx1250 async copy: global -> LDS, 16 bytes per lane, tracked by ASYNCcnt.
static __device__ __forceinline__ void async_b128(unsigned lds_off, const void* gptr) {
  asm volatile("global_load_async_to_lds_b128 %0, %1, off"
               :: "v"(lds_off), "v"(gptr) : "memory");
}
static __device__ __forceinline__ void wait_async0() {
  asm volatile("s_wait_asynccnt 0x0" ::: "memory");
}

// A fragment (16x32 bf16). lane m=l&15, hi=l>>4.
// element run [0..7] = K[8*hi .. 8*hi+7], run [8..15] = K[16+8*hi .. +7].
static __device__ __forceinline__ v16bf frag_A(const unsigned short* sA, int mb, int lane) {
  int m  = mb + (lane & 15);
  int hi = (lane >> 4) & 1;
  const unsigned short* p = sA + m * AS + 8 * hi;
  uint4 lo = *(const uint4*)(p);
  uint4 hh = *(const uint4*)(p + 16);
  v8u pk = {lo.x, lo.y, lo.z, lo.w, hh.x, hh.y, hh.z, hh.w};
  return __builtin_bit_cast(v16bf, pk);
}

// B fragment (32x16 bf16), tile stored [n][k] (k-contiguous).
// lane n=l&15, hi=l>>4: run [0..7] = K[8*hi .. 8*hi+7], run [8..15] = K[16+8*hi .. +7].
static __device__ __forceinline__ v16bf frag_B(const unsigned short* sB, int nb, int lane) {
  int n  = nb + (lane & 15);
  int hi = (lane >> 4) & 1;
  const unsigned short* p = sB + n * BS + 8 * hi;
  uint4 lo = *(const uint4*)(p);
  uint4 hh = *(const uint4*)(p + 16);
  v8u pk = {lo.x, lo.y, lo.z, lo.w, hh.x, hh.y, hh.z, hh.w};
  return __builtin_bit_cast(v16bf, pk);
}

// ---------------------------------------------------------------------------
// Kernel 0: build transposed bf16 weights, zero ping-pong h buffers.
//   WhhT: 2 dir x (n-major HxH)   WxhT: 2 dir x (n-major HxD)
// Reads are coalesced (n contiguous); one-time scattered u16 writes.
// ---------------------------------------------------------------------------
__global__ __launch_bounds__(256) void prep_kernel(
    const float* __restrict__ fWxh, const float* __restrict__ fWhh,
    const float* __restrict__ bWxh, const float* __restrict__ bWhh,
    unsigned short* __restrict__ WhhT, unsigned short* __restrict__ WxhT,
    unsigned short* __restrict__ hbf) {
  int i = blockIdx.x * blockDim.x + threadIdx.x;
  const int nWhh = HH * HH;
  const int nWxh = DD * HH;
  if (i < 2 * nWhh) {
    int d = i / nWhh, r = i - d * nWhh;
    int k = r / HH, n = r - k * HH;                  // consecutive i -> consecutive n
    const float* W = d ? bWhh : fWhh;
    WhhT[(size_t)d * nWhh + (size_t)n * HH + k] = f2bf(W[(size_t)k * HH + n]);
  } else if (i < 2 * nWhh + 2 * nWxh) {
    int j = i - 2 * nWhh;
    int d = j / nWxh, r = j - d * nWxh;
    int k = r / HH, n = r - k * HH;
    const float* W = d ? bWxh : fWxh;
    WxhT[(size_t)d * nWxh + (size_t)n * DD + k] = f2bf(W[(size_t)k * HH + n]);
  }
  if (i < 4 * BB * HH) hbf[i] = 0;                   // h0 = 0
}

// ---------------------------------------------------------------------------
// Kernel 1: X = inputs @ Wxh + bh for both directions, staged into d_out.
// Grid: wg = dir + 2*ntile + 16*mtile ; 2 dirs * 8 ntiles(128) * 512 mtiles(64)
// ---------------------------------------------------------------------------
__global__ __launch_bounds__(256) void xw_gemm_kernel(
    const float* __restrict__ x,
    const unsigned short* __restrict__ WxhT,  // 2 x (H x D) bf16, k-contiguous
    const float* __restrict__ fb, const float* __restrict__ bb,
    float* __restrict__ out) {
  __shared__ __align__(16) unsigned short sA[64 * AS];
  __shared__ __align__(16) unsigned short sB[128 * BS];

  const int wg    = blockIdx.x;
  const int dir   = wg & 1;
  const int ntile = (wg >> 1) & 7;
  const int mtile = wg >> 4;
  const unsigned short* __restrict__ WT = WxhT + (size_t)dir * DD * HH;
  const float* __restrict__ bias        = dir ? bb : fb;

  const int tid  = threadIdx.x;
  const int lane = tid & 31;
  const int wv   = tid >> 5;
  const int mi   = wv & 3;        // 16-row subtile
  const int ng   = wv >> 2;       // 64-col group
  const int m0   = mtile * 64;
  const int n0   = ntile * 128;

  const unsigned sAoff = (unsigned)(size_t)(void*)sA;
  const unsigned sBoff = (unsigned)(size_t)(void*)sB;

  v8f acc[4] = {v8f{}, v8f{}, v8f{}, v8f{}};

  for (int k0 = 0; k0 < DD; k0 += 32) {
    // B tile: pure bf16 copy -> async to LDS (2 x b128 per thread)
#pragma unroll
    for (int it = 0; it < 2; ++it) {
      int i  = tid + it * 256;          // 0..511
      int n  = i >> 2;                  // 128 rows
      int sg = i & 3;                   // 4 x 16B per 64B row
      async_b128(sBoff + (unsigned)(n * BS + sg * 8) * 2,
                 WT + (size_t)(n0 + n) * DD + k0 + sg * 8);
    }
    // A tile: f32 -> bf16 convert (2 x float4 per thread)
#pragma unroll
    for (int it = 0; it < 2; ++it) {
      int i  = tid + it * 256;          // 0..511
      int m  = i >> 3;                  // 64 rows, 8 float4 per row
      int sg = i & 7;
      float4 g = *(const float4*)(x + (size_t)(m0 + m) * DD + k0 + sg * 4);
      uint2 p;
      p.x = (unsigned)f2bf(g.x) | ((unsigned)f2bf(g.y) << 16);
      p.y = (unsigned)f2bf(g.z) | ((unsigned)f2bf(g.w) << 16);
      *(uint2*)(sA + m * AS + sg * 4) = p;
    }
    wait_async0();
    __syncthreads();

    v16bf a = frag_A(sA, mi * 16, lane);
#pragma unroll
    for (int j = 0; j < 4; ++j) {
      v16bf b = frag_B(sB, ng * 64 + j * 16, lane);
      acc[j] = __builtin_amdgcn_wmma_f32_16x16x32_bf16(
          false, a, false, b, (short)0, acc[j], false, false);
    }
    __syncthreads();
  }

  const int hi = (lane >> 4) & 1;
#pragma unroll
  for (int j = 0; j < 4; ++j) {
    int n = n0 + ng * 64 + j * 16 + (lane & 15);
    float bv = bias[n];
#pragma unroll
    for (int r = 0; r < 8; ++r) {
      int m = m0 + mi * 16 + hi * 8 + r;
      out[(size_t)m * (2 * HH) + (size_t)dir * HH + n] = acc[j][r] + bv;
    }
  }
}

// ---------------------------------------------------------------------------
// Kernel 2: one recurrence step for both directions (16 WGs).
//   h_new = tanh(X[tslot] + h_prev @ Whh)  (X in-place in d_out)
// ---------------------------------------------------------------------------
__global__ __launch_bounds__(256) void rnn_step_kernel(
    const unsigned short* __restrict__ WhhT,  // 2 x (H x H) bf16, k-contiguous
    unsigned short* __restrict__ hbf,         // 2 dir x 2 phase x (B*H) bf16
    float* __restrict__ out,
    int s) {
  __shared__ __align__(16) unsigned short sA[64 * AS];
  __shared__ __align__(16) unsigned short sB[128 * BS];

  const int wg    = blockIdx.x;
  const int dir   = wg & 1;
  const int ntile = wg >> 1;
  const int tslot = dir ? (T_SEQ - 1 - s) : s;

  const unsigned short* __restrict__ WT = WhhT + (size_t)dir * HH * HH;
  const unsigned short* __restrict__ hprev =
      hbf + ((size_t)dir * 2 + (s & 1)) * (BB * HH);
  unsigned short* __restrict__ hnext =
      hbf + ((size_t)dir * 2 + ((s + 1) & 1)) * (BB * HH);

  const int tid  = threadIdx.x;
  const int lane = tid & 31;
  const int wv   = tid >> 5;
  const int mi   = wv & 3;
  const int ng   = wv >> 2;
  const int n0   = ntile * 128;

  const unsigned sAoff = (unsigned)(size_t)(void*)sA;
  const unsigned sBoff = (unsigned)(size_t)(void*)sB;

  v8f acc[4] = {v8f{}, v8f{}, v8f{}, v8f{}};

  for (int k0 = 0; k0 < HH; k0 += 32) {
    // A tile: bf16 h_prev copy, 1 x b128 per thread (64 rows x 4 segs)
    {
      int m  = tid >> 2;
      int sg = tid & 3;
      async_b128(sAoff + (unsigned)(m * AS + sg * 8) * 2,
                 hprev + (size_t)m * HH + k0 + sg * 8);
    }
    // B tile: bf16 WhhT copy, 2 x b128 per thread
#pragma unroll
    for (int it = 0; it < 2; ++it) {
      int i  = tid + it * 256;
      int n  = i >> 2;
      int sg = i & 3;
      async_b128(sBoff + (unsigned)(n * BS + sg * 8) * 2,
                 WT + (size_t)(n0 + n) * HH + k0 + sg * 8);
    }
    wait_async0();
    __syncthreads();

    v16bf a = frag_A(sA, mi * 16, lane);
#pragma unroll
    for (int j = 0; j < 4; ++j) {
      v16bf b = frag_B(sB, ng * 64 + j * 16, lane);
      acc[j] = __builtin_amdgcn_wmma_f32_16x16x32_bf16(
          false, a, false, b, (short)0, acc[j], false, false);
    }
    __syncthreads();
  }

  const int hi = (lane >> 4) & 1;
#pragma unroll
  for (int j = 0; j < 4; ++j) {
    int n = n0 + ng * 64 + j * 16 + (lane & 15);
#pragma unroll
    for (int r = 0; r < 8; ++r) {
      int b = mi * 16 + hi * 8 + r;   // batch row
      size_t o = ((size_t)tslot * BB + b) * (2 * HH) + (size_t)dir * HH + n;
      float v = tanhf(acc[j][r] + out[o]);
      out[o] = v;
      hnext[(size_t)b * HH + n] = f2bf(v);
    }
  }
}

// ---------------------------------------------------------------------------
// Kernel 3: copy final states into the tail of d_out.
// ---------------------------------------------------------------------------
__global__ __launch_bounds__(256) void finalize_kernel(float* __restrict__ out) {
  int i = blockIdx.x * blockDim.x + threadIdx.x;
  const int half = BB * HH;
  float* tail = out + (size_t)T_SEQ * BB * 2 * HH;
  if (i < half) {
    int b = i / HH, n = i % HH;
    tail[i] = out[((size_t)(T_SEQ - 1) * BB + b) * (2 * HH) + n];       // f_H
  } else if (i < 2 * half) {
    int j = i - half;
    int b = j / HH, n = j % HH;
    tail[i] = out[((size_t)b) * (2 * HH) + HH + n];                     // b_H
  }
}

// ---------------------------------------------------------------------------
extern "C" void kernel_launch(void* const* d_in, const int* in_sizes, int n_in,
                              void* d_out, int out_size, void* d_ws, size_t ws_size,
                              hipStream_t stream) {
  (void)in_sizes; (void)n_in; (void)out_size; (void)ws_size;
  const float* x    = (const float*)d_in[0];  // (T,B,D)
  const float* fWxh = (const float*)d_in[1];  // (D,H)
  const float* fWhh = (const float*)d_in[2];  // (H,H)
  const float* fbh  = (const float*)d_in[3];  // (H,)
  const float* bWxh = (const float*)d_in[4];
  const float* bWhh = (const float*)d_in[5];
  const float* bbh  = (const float*)d_in[6];
  float* out = (float*)d_out;

  // Workspace: [ WhhT 2x1Mx2B | WxhT 2x512Kx2B | h ping-pong 4x64Kx2B ] ~6.5MB
  unsigned short* WhhT = (unsigned short*)d_ws;
  unsigned short* WxhT = WhhT + (size_t)2 * HH * HH;
  unsigned short* hbf  = WxhT + (size_t)2 * DD * HH;

  // 0) weight transpose/convert + h0 = 0
  const int prepN = 2 * HH * HH + 2 * DD * HH;
  prep_kernel<<<(prepN + 255) / 256, 256, 0, stream>>>(
      fWxh, fWhh, bWxh, bWhh, WhhT, WxhT, hbf);

  // 1) input-projection GEMM (both directions) staged into d_out
  xw_gemm_kernel<<<(T_SEQ * BB / 64) * 8 * 2, 256, 0, stream>>>(
      x, WxhT, fbh, bbh, out);

  // 2) serial recurrence: 512 steps, fwd+bwd fused per launch
  for (int s = 0; s < T_SEQ; ++s)
    rnn_step_kernel<<<16, 256, 0, stream>>>(WhhT, hbf, out, s);

  // 3) final hidden states into the output tail
  finalize_kernel<<<(2 * BB * HH + 255) / 256, 256, 0, stream>>>(out);
}
